// TGATLayer_64707977282178
// MI455X (gfx1250) — matrix operand order repaired and checked by reading, not compiled, not run on hardware
//
#include <hip/hip_runtime.h>
#include <hip/hip_bf16.h>

// ---------------------------------------------------------------------------
// TGAT layer for gfx1250 (MI455X), wave32 + WMMA, async-to-LDS staged weights.
// N=100000 nodes, E=1600000 edges, IN_C=128, OUT_C=16, TIME_C=64, HEADS=4, HID=64
// ---------------------------------------------------------------------------

typedef __attribute__((ext_vector_type(16))) _Float16 v16h;
typedef __attribute__((ext_vector_type(8)))  float    v8f;
typedef __attribute__((ext_vector_type(2)))  float    v2f;

#define IN_C   128
#define OUT_C  16
#define TIME_C 64
#define HEADS  4
#define HID    64   // HEADS*OUT_C

// ---- WMMA helpers ---------------------------------------------------------

__device__ __forceinline__ v8f wmma_f16(v16h a, v16h b, v8f c) {
  // D(16x16,f32) = A(16x32,f16) * B(32x16,f16) + C
  return __builtin_amdgcn_wmma_f32_16x16x32_f16(false, a, false, b, (short)0, c,
                                                false, false);
}

// Monotone float <-> uint transform so atomicMax(uint) == float max.
__device__ __forceinline__ unsigned f2mono(float f) {
  unsigned u = __float_as_uint(f);
  return (u & 0x80000000u) ? ~u : (u | 0x80000000u);
}
__device__ __forceinline__ float mono2f(unsigned u) {
  unsigned b = (u & 0x80000000u) ? (u & 0x7FFFFFFFu) : ~u;
  return __uint_as_float(b);
}

// A-fragment slot -> K mapping for f16 16x32 A tiles (ISA 7.12.2).
__device__ __forceinline__ int a_kmap(int h, int j) {
  return (j < 8) ? (h * 8 + j) : (16 + h * 8 + (j - 8));
}

// ---- Kernel 0a: pack f32 weights into f16 WMMA B-fragment layout ----------
// P[(((t*nChunks + c)*32 + lane)*16) + j] = W[(c*32 + a_kmap(h, j))*ncols + t*16 + n]
// so a wave reads its whole B fragment as one contiguous 32B per lane.

__global__ void k_pack_f16(const float* __restrict__ W, int ncols, int nChunks,
                           _Float16* __restrict__ P, int total) {
  int i = blockIdx.x * blockDim.x + threadIdx.x;
  if (i >= total) return;
  int j = i & 15;
  int lane = (i >> 4) & 31;
  int rest = i >> 9;
  int c = rest % nChunks;
  int t = rest / nChunks;
  int n = lane & 15, h = lane >> 4;
  int k = c * 32 + a_kmap(h, j);
  P[i] = (_Float16)W[k * ncols + t * 16 + n];
}

// Pack Wout (64x16 f32) into f32 16x16x4 B-fragment layout:
// P[(kb*32 + lane)*2 + vg] = Wout[(kb*4 + h*2 + vg)*16 + n]
__global__ void k_pack_wout(const float* __restrict__ W, float* __restrict__ P) {
  int i = blockIdx.x * blockDim.x + threadIdx.x;
  if (i >= (HID / 4) * 32 * 2) return;
  int vg = i & 1;
  int lane = (i >> 1) & 31;
  int kb = i >> 6;
  int n = lane & 15, h = lane >> 4;
  P[i] = W[(kb * 4 + h * 2 + vg) * OUT_C + n];
}

// ---- Kernel 0b: zero-init accumulators ------------------------------------

__global__ void k_init(unsigned* __restrict__ mu, float* __restrict__ den,
                       float* __restrict__ agg, int nNodes) {
  int i = blockIdx.x * blockDim.x + threadIdx.x;
  int stride = gridDim.x * blockDim.x;
  for (int j = i; j < nNodes * HEADS; j += stride) {
    mu[j] = 0u;  // below f2mono of any finite score: identity for max
    den[j] = 0.0f;
  }
  for (int j = i; j < nNodes * HID; j += stride) agg[j] = 0.0f;
}

// ---- async global->LDS staging (16B per lane per op, ASYNCcnt-tracked) ----
// GVS mode: mem_addr = SADDR(64b SGPR) + VADDR(32b byte offset); LDS dest =
// LDS_BASE + VDST. Low 32 bits of a flat LDS pointer are the LDS byte offset
// (ISA 10.2 aperture mapping).

__device__ __forceinline__ void cp_lds_async(_Float16* dstLds,
                                             const _Float16* src, int nHalves,
                                             int tid, int nt) {
  unsigned ldsBase = (unsigned)(size_t)dstLds;
  int n16 = nHalves >> 3;  // 16-byte packets
  for (int i = tid; i < n16; i += nt) {
    unsigned dsa = ldsBase + (unsigned)i * 16u;
    int goff = i * 16;
    asm volatile("global_load_async_to_lds_b128 %0, %1, %2"
                 :
                 : "v"(dsa), "v"(goff), "s"(src)
                 : "memory");
  }
  asm volatile("s_wait_asynccnt 0x0" ::: "memory");
}

// ---- 16x16 output tile GEMM from LDS-staged packed fragments --------------

__device__ __forceinline__ void gemm_tile_lds(const v16h* a, int nChunks,
                                              const _Float16* sW, int t,
                                              int lane, float* __restrict__ dst,
                                              int ldD, int colbase, int rowbase) {
  int n = lane & 15, h = lane >> 4;
  v8f acc = {};
  for (int c = 0; c < nChunks; ++c) {
    v16h b = *(const v16h*)(sW + ((t * nChunks + c) * 32 + lane) * 16);
    acc = wmma_f16(a[c], b, acc);
  }
#pragma unroll
  for (int v = 0; v < 8; ++v) {
    int mr = v + 8 * h;  // D layout: lane=N, VGPR v -> M = v + 8h
    dst[(rowbase + mr) * ldD + colbase + n] = acc[v];
  }
}

// ---- Kernel 1: q/k/v + x@Wroot node projections (one wave per 16 nodes) ---

__global__ void k_node_proj(const float* __restrict__ x,
                            const _Float16* __restrict__ PWq,
                            const _Float16* __restrict__ PWk,
                            const _Float16* __restrict__ PWv,
                            const _Float16* __restrict__ PWroot,
                            float* __restrict__ qf, float* __restrict__ kf,
                            float* __restrict__ vf, float* __restrict__ xr,
                            int nNodes) {
  // packed fragment sizes (halves): Wq/Wk/Wv 4*4*512=8192 each, Wroot 2048
  __shared__ __align__(16) _Float16 sW[8192 * 3 + 2048];
  _Float16* sWq = sW;
  _Float16* sWk = sW + 8192;
  _Float16* sWv = sW + 16384;
  _Float16* sWroot = sW + 24576;
  int tid = threadIdx.x, nt = blockDim.x;
  cp_lds_async(sWq, PWq, 8192, tid, nt);
  cp_lds_async(sWk, PWk, 8192, tid, nt);
  cp_lds_async(sWv, PWv, 8192, tid, nt);
  cp_lds_async(sWroot, PWroot, 2048, tid, nt);
  __syncthreads();

  int lane = threadIdx.x & 31;
  int tile = (blockIdx.x * blockDim.x + threadIdx.x) >> 5;
  if (tile * 16 >= nNodes) return;  // wave-uniform: EXEC all-ones at WMMA
  int m = lane & 15, h = lane >> 4;
  int node = tile * 16 + m;
  if (node >= nNodes) node = nNodes - 1;

  // A = x tile 16x128 f16: per chunk each lane owns two runs of 8 floats.
  v16h a[4];
#pragma unroll
  for (int c = 0; c < 4; ++c) {
    const float* base = x + node * IN_C + c * 32 + h * 8;
    float4 f0 = *(const float4*)(base);
    float4 f1 = *(const float4*)(base + 4);
    float4 f2 = *(const float4*)(base + 16);
    float4 f3 = *(const float4*)(base + 20);
    a[c][0] = (_Float16)f0.x;  a[c][1] = (_Float16)f0.y;
    a[c][2] = (_Float16)f0.z;  a[c][3] = (_Float16)f0.w;
    a[c][4] = (_Float16)f1.x;  a[c][5] = (_Float16)f1.y;
    a[c][6] = (_Float16)f1.z;  a[c][7] = (_Float16)f1.w;
    a[c][8] = (_Float16)f2.x;  a[c][9] = (_Float16)f2.y;
    a[c][10] = (_Float16)f2.z; a[c][11] = (_Float16)f2.w;
    a[c][12] = (_Float16)f3.x; a[c][13] = (_Float16)f3.y;
    a[c][14] = (_Float16)f3.z; a[c][15] = (_Float16)f3.w;
  }

  for (int t = 0; t < HEADS; ++t) {
    gemm_tile_lds(a, 4, sWq, t, lane, qf, HID, t * 16, tile * 16);
    gemm_tile_lds(a, 4, sWk, t, lane, kf, HID, t * 16, tile * 16);
    gemm_tile_lds(a, 4, sWv, t, lane, vf, HID, t * 16, tile * 16);
  }
  gemm_tile_lds(a, 4, sWroot, 0, lane, xr, OUT_C, 0, tile * 16);
}

// ---- time-encoding A fragments: te = cos(delta*w + b), 16 edges x 64 ------

__device__ __forceinline__ void build_te_frags(float delta,
                                               const float* __restrict__ tw,
                                               const float* __restrict__ tb,
                                               int h, v16h a[2]) {
#pragma unroll
  for (int c = 0; c < 2; ++c)
#pragma unroll
    for (int j = 0; j < 16; ++j) {
      int k = c * 32 + a_kmap(h, j);
      a[c][j] = (_Float16)__cosf(delta * tw[k] + tb[k]);
    }
}

// ---- Kernel 2: per-edge scores + segment max (one wave per 16 edges) ------

__global__ void k_edge_scores(const int* __restrict__ edge_index,
                              const float* __restrict__ edge_time,
                              const float* __restrict__ query_time,
                              const float* __restrict__ time_w,
                              const float* __restrict__ time_b,
                              const _Float16* __restrict__ PWt,
                              const float* __restrict__ qf,
                              const float* __restrict__ kf,
                              float* __restrict__ sc, unsigned* __restrict__ mu,
                              int nEdges) {
  __shared__ __align__(16) _Float16 sWt[8192];  // 4 tiles * 2 chunks * 512
  cp_lds_async(sWt, PWt, 8192, threadIdx.x, blockDim.x);
  __syncthreads();

  int lane = threadIdx.x & 31;
  int tile = (blockIdx.x * blockDim.x + threadIdx.x) >> 5;
  if (tile * 16 >= nEdges) return;
  int m = lane & 15, h = lane >> 4, n = m;

  int e = tile * 16 + m;
  float delta = 0.0f;
  if (e < nEdges) {
    int cIdx = edge_index[nEdges + e];  // col = destination
    delta = query_time[cIdx] - edge_time[e];
  }
  v16h a[2];
  build_te_frags(delta, time_w, time_b, h, a);

  int cr[8], rr[8];
  bool evr[8];
#pragma unroll
  for (int v = 0; v < 8; ++v) {
    int er = tile * 16 + 8 * h + v;  // edges owned by this lane's D-rows
    evr[v] = er < nEdges;
    cr[v] = evr[v] ? edge_index[nEdges + er] : 0;
    rr[v] = evr[v] ? edge_index[er] : 0;
  }

  for (int t = 0; t < HEADS; ++t) {  // n-tile t == head t
    v8f acc = {};
#pragma unroll
    for (int c = 0; c < 2; ++c) {
      v16h b = *(const v16h*)(sWt + ((t * 2 + c) * 32 + lane) * 16);
      acc = wmma_f16(a[c], b, acc);  // tb tile: 16 edges x 16 cols
    }
#pragma unroll
    for (int v = 0; v < 8; ++v) {
      float p = 0.0f;
      if (evr[v])
        p = qf[cr[v] * HID + t * 16 + n] *
            (kf[rr[v] * HID + t * 16 + n] + acc[v]);
      // butterfly sum over each 16-lane half
      p += __shfl_xor(p, 1);
      p += __shfl_xor(p, 2);
      p += __shfl_xor(p, 4);
      p += __shfl_xor(p, 8);
      if (n == 0 && evr[v]) {
        int er = tile * 16 + 8 * h + v;
        float s = p * 0.25f;  // 1/sqrt(OUT_C)
        sc[er * HEADS + t] = s;
        atomicMax(&mu[cr[v] * HEADS + t], f2mono(s));
      }
    }
  }
}

// ---- Kernel 3: softmax denominator ----------------------------------------

__global__ void k_denom(const int* __restrict__ edge_index,
                        const float* __restrict__ sc,
                        const unsigned* __restrict__ mu,
                        float* __restrict__ den, int nEdges) {
  int i = blockIdx.x * blockDim.x + threadIdx.x;
  if (i >= nEdges * HEADS) return;
  int e = i >> 2, t = i & 3;
  int c = edge_index[nEdges + e];
  float mmax = mono2f(mu[c * HEADS + t]);
  atomicAdd(&den[c * HEADS + t], __expf(sc[i] - mmax));
}

// ---- Kernel 4: weighted message scatter (recompute tb via WMMA) -----------

__global__ void k_edge_msg(const int* __restrict__ edge_index,
                           const float* __restrict__ edge_time,
                           const float* __restrict__ query_time,
                           const float* __restrict__ time_w,
                           const float* __restrict__ time_b,
                           const _Float16* __restrict__ PWt,
                           const float* __restrict__ vf,
                           const float* __restrict__ sc,
                           const unsigned* __restrict__ mu,
                           const float* __restrict__ den,
                           float* __restrict__ agg, int nEdges) {
  __shared__ __align__(16) _Float16 sWt[8192];
  cp_lds_async(sWt, PWt, 8192, threadIdx.x, blockDim.x);
  __syncthreads();

  int lane = threadIdx.x & 31;
  int tile = (blockIdx.x * blockDim.x + threadIdx.x) >> 5;
  if (tile * 16 >= nEdges) return;
  int m = lane & 15, h = lane >> 4, n = m;

  int e = tile * 16 + m;
  float delta = 0.0f;
  if (e < nEdges) {
    int cIdx = edge_index[nEdges + e];
    delta = query_time[cIdx] - edge_time[e];
  }
  v16h a[2];
  build_te_frags(delta, time_w, time_b, h, a);

  int cr[8], rr[8];
  bool evr[8];
#pragma unroll
  for (int v = 0; v < 8; ++v) {
    int er = tile * 16 + 8 * h + v;
    evr[v] = er < nEdges;
    cr[v] = evr[v] ? edge_index[nEdges + er] : 0;
    rr[v] = evr[v] ? edge_index[er] : 0;
  }

  for (int t = 0; t < HEADS; ++t) {
    v8f acc = {};
#pragma unroll
    for (int c = 0; c < 2; ++c) {
      v16h b = *(const v16h*)(sWt + ((t * 2 + c) * 32 + lane) * 16);
      acc = wmma_f16(a[c], b, acc);
    }
#pragma unroll
    for (int v = 0; v < 8; ++v) {
      if (!evr[v]) continue;
      int er = tile * 16 + 8 * h + v;
      float mmax = mono2f(mu[cr[v] * HEADS + t]);
      float dd = den[cr[v] * HEADS + t];
      float w = __expf(sc[er * HEADS + t] - mmax) / dd;
      float msg = (vf[rr[v] * HID + t * 16 + n] + acc[v]) * w;
      atomicAdd(&agg[cr[v] * HID + t * 16 + n], msg);
    }
  }
}

// ---- Kernel 5: out = agg@Wout + bout + x@Wroot  (fp32 WMMA 16x16x4) -------

__global__ void k_out(const float* __restrict__ agg,
                      const float* __restrict__ PWout,
                      const float* __restrict__ bout,
                      const float* __restrict__ xrp, float* __restrict__ out,
                      int nNodes) {
  int lane = threadIdx.x & 31;
  int tile = (blockIdx.x * blockDim.x + threadIdx.x) >> 5;
  if (tile * 16 >= nNodes) return;
  int m = lane & 15, h = lane >> 4, n = m;
  int node = tile * 16 + m;
  if (node >= nNodes) node = nNodes - 1;

  v8f acc = {};
#pragma unroll
  for (int kb = 0; kb < HID / 4; ++kb) {
    // f32 A 16x4: lane m holds K = h*2 + vg
    v2f a2 = *(const v2f*)(agg + node * HID + kb * 4 + h * 2);
    v2f b2 = *(const v2f*)(PWout + (kb * 32 + lane) * 2);
    acc = __builtin_amdgcn_wmma_f32_16x16x4_f32(false, a2, false, b2, (short)0,
                                                acc, false, false);
  }
#pragma unroll
  for (int v = 0; v < 8; ++v) {
    int nd = tile * 16 + v + 8 * h;
    if (nd < nNodes) out[nd * OUT_C + n] = acc[v] + bout[n] + xrp[nd * OUT_C + n];
  }
}

// ---------------------------------------------------------------------------

extern "C" void kernel_launch(void* const* d_in, const int* in_sizes, int n_in,
                              void* d_out, int out_size, void* d_ws,
                              size_t ws_size, hipStream_t stream) {
  const float* x          = (const float*)d_in[0];
  const int*   edge_index = (const int*)d_in[1];
  const float* edge_time  = (const float*)d_in[2];
  const float* query_time = (const float*)d_in[3];
  const float* Wq         = (const float*)d_in[4];
  const float* Wk         = (const float*)d_in[5];
  const float* Wv         = (const float*)d_in[6];
  const float* time_w     = (const float*)d_in[7];
  const float* time_b     = (const float*)d_in[8];
  const float* Wt         = (const float*)d_in[9];
  const float* Wout       = (const float*)d_in[10];
  const float* bout       = (const float*)d_in[11];
  const float* Wroot      = (const float*)d_in[12];
  float* out = (float*)d_out;

  const int N = in_sizes[0] / IN_C;
  const int E = in_sizes[2];

  char* ws = (char*)d_ws;
  size_t off = 0;
  auto carve = [&](size_t bytes) -> void* {
    void* p = ws + off;
    off += (bytes + 255) & ~(size_t)255;
    return p;
  };
  float*    qf     = (float*)carve((size_t)N * HID * 4);
  float*    kf     = (float*)carve((size_t)N * HID * 4);
  float*    vf     = (float*)carve((size_t)N * HID * 4);
  float*    xrp    = (float*)carve((size_t)N * OUT_C * 4);
  float*    agg    = (float*)carve((size_t)N * HID * 4);
  unsigned* mu     = (unsigned*)carve((size_t)N * HEADS * 4);
  float*    den    = (float*)carve((size_t)N * HEADS * 4);
  float*    sc     = (float*)carve((size_t)E * HEADS * 4);
  _Float16* PWq    = (_Float16*)carve(8192 * 2);
  _Float16* PWk    = (_Float16*)carve(8192 * 2);
  _Float16* PWv    = (_Float16*)carve(8192 * 2);
  _Float16* PWt    = (_Float16*)carve(4096 * 2);
  _Float16* PWroot = (_Float16*)carve(2048 * 2);
  float*    PWout  = (float*)carve(1024 * 4);

  // 0) weight packing into WMMA fragment layouts + accumulator init
  k_pack_f16<<<(8192 + 255) / 256, 256, 0, stream>>>(Wq, HID, 4, PWq, 8192);
  k_pack_f16<<<(8192 + 255) / 256, 256, 0, stream>>>(Wk, HID, 4, PWk, 8192);
  k_pack_f16<<<(8192 + 255) / 256, 256, 0, stream>>>(Wv, HID, 4, PWv, 8192);
  k_pack_f16<<<(4096 + 255) / 256, 256, 0, stream>>>(Wt, HID, 2, PWt, 4096);
  k_pack_f16<<<(2048 + 255) / 256, 256, 0, stream>>>(Wroot, OUT_C, 4, PWroot, 2048);
  k_pack_wout<<<(1024 + 255) / 256, 256, 0, stream>>>(Wout, PWout);
  k_init<<<1024, 256, 0, stream>>>(mu, den, agg, N);

  // 1) node projections (one wave per 16-node tile; 4 waves per block)
  int nodeTiles = (N + 15) / 16;
  k_node_proj<<<(nodeTiles + 3) / 4, 128, 0, stream>>>(x, PWq, PWk, PWv, PWroot,
                                                       qf, kf, vf, xrp, N);

  // 2) per-edge scores + segment max
  int edgeTiles = (E + 15) / 16;
  k_edge_scores<<<(edgeTiles + 3) / 4, 128, 0, stream>>>(
      edge_index, edge_time, query_time, time_w, time_b, PWt, qf, kf, sc, mu, E);

  // 3) softmax denominators
  k_denom<<<(E * HEADS + 255) / 256, 256, 0, stream>>>(edge_index, sc, mu, den, E);

  // 4) weighted message scatter-add
  k_edge_msg<<<(edgeTiles + 3) / 4, 128, 0, stream>>>(
      edge_index, edge_time, query_time, time_w, time_b, PWt, vf, sc, mu, den,
      agg, E);

  // 5) output projection (fp32 WMMA) + skip connection
  k_out<<<(nodeTiles + 3) / 4, 128, 0, stream>>>(agg, PWout, bout, xrp, out, N);
}
